// VanillaConvolutionWrapper_55284819034311
// MI455X (gfx1250) — matrix operand order repaired and checked by reading, not compiled, not run on hardware
//
#include <hip/hip_runtime.h>

// ---------------------------------------------------------------------------
// Conv2d 3x3 s1 p1 as implicit GEMM on CDNA5 WMMA (3xBF16 split, f32 accum)
//   M = COUT = 256, N = B*H*W = 50176, K = CIN*9 = 1152  (k = tap*128 + ci)
// Workgroup: 256 threads = 8 waves (2 M x 4 N), tile 64x128, K-step 32.
// ---------------------------------------------------------------------------

typedef __attribute__((ext_vector_type(16))) __bf16 v16bf;
typedef __attribute__((ext_vector_type(8)))  float  v8f;

#define CIN   128
#define COUT  256
#define HW    3136          // 56*56
#define KTOT  1152
#define TILE_M 64
#define TILE_N 128
#define AS_STRIDE 40        // bf16 units per A row (32 + 8 pad)
#define BS_STRIDE 40        // bf16 units per B row (32 + 8 pad)

union BfPack { unsigned int u; __bf16 h[2]; };

__device__ __forceinline__ unsigned int pack2(__bf16 a, __bf16 b) {
    BfPack p; p.h[0] = a; p.h[1] = b; return p.u;
}

__device__ __forceinline__ void split_bf16(float v, __bf16& hi, __bf16& lo) {
    hi = (__bf16)v;
    lo = (__bf16)(v - (float)hi);
}

__device__ __forceinline__ v16bf load_frag(const __bf16* p) {
    v16bf r;
    ((float4*)&r)[0] = ((const float4*)p)[0];   // ds_load_b128
    ((float4*)&r)[1] = ((const float4*)p)[1];   // ds_load_b128
    return r;
}

__global__ __launch_bounds__(256)
void conv_wmma_kernel(const float* __restrict__ x, const float* __restrict__ w,
                      const float* __restrict__ bias, float* __restrict__ out) {
    __shared__ __bf16 AsH[TILE_M * AS_STRIDE];
    __shared__ __bf16 AsL[TILE_M * AS_STRIDE];
    __shared__ __bf16 BsH[TILE_N * BS_STRIDE];
    __shared__ __bf16 BsL[TILE_N * BS_STRIDE];

    const int t    = threadIdx.x;
    const int lane = t & 31;
    const int wave = t >> 5;
    const int half = lane >> 4;
    const int l16  = lane & 15;
    const int wm   = wave >> 2;      // 0..1  (M sub-block of 32)
    const int wn   = wave & 3;       // 0..3  (N sub-block of 32)

    const int m0 = blockIdx.y * TILE_M;
    const int n0 = blockIdx.x * TILE_N;

    // --- A (weights) load assignment: 64 rows x 32 k, 8 elems/thread ------
    const int am  = t >> 2;            // 0..63
    const int ak0 = (t & 3) * 8;       // 8 consecutive klocal

    // --- B (image gather) assignment: 128 n x 32 k, 16 elems/thread ------
    const int bn   = t & 127;
    const int bk0  = (t >> 7) * 16;
    const int ng_b = n0 + bn;
    const int bb   = ng_b / HW;
    const int hwb  = ng_b - bb * HW;
    const int hb   = hwb / 56;
    const int wb   = hwb - hb * 56;

    v8f acc[2][2] = {};

    for (int it = 0; it < 36; ++it) {
        const int tap = it >> 2;              // filter tap 0..8 (uniform)
        const int cib = (it & 3) * 32;        // ci base of this K-slice
        const int kh  = tap / 3;
        const int kw  = tap - kh * 3;

        // ---------------- global loads (overlap previous compute) --------
        float areg[8];
        const float* wp = w + (m0 + am) * KTOT + (cib + ak0) * 9 + tap;
        #pragma unroll
        for (int j = 0; j < 8; ++j) areg[j] = wp[j * 9];

        const int ih = hb + kh - 1;
        const int iw = wb + kw - 1;
        const bool inb = ((unsigned)ih < 56u) && ((unsigned)iw < 56u);
        const float* xp = x + (((bb * CIN + cib + bk0) * 56 + ih) * 56 + iw);
        float breg[16];
        #pragma unroll
        for (int j = 0; j < 16; ++j) breg[j] = inb ? xp[j * HW] : 0.0f;

        if (it + 1 < 36) {                    // hint next weight tile
            __builtin_prefetch(wp + 32 * 9, 0, 3);
        }

        __syncthreads();                      // previous compute done reading

        // ---------------- stage A into LDS in WMMA fragment order --------
        #pragma unroll
        for (int p = 0; p < 4; ++p) {
            const int kl = ak0 + 2 * p;       // even klocal
            const int hh = (kl >> 3) & 1;
            const int vv = ((kl & 16) ? 4 : 0) + ((kl & 7) >> 1);
            const int pos = hh * 16 + vv * 2;
            __bf16 h0, l0, h1, l1;
            split_bf16(areg[2 * p], h0, l0);
            split_bf16(areg[2 * p + 1], h1, l1);
            *(unsigned int*)&AsH[am * AS_STRIDE + pos] = pack2(h0, h1);
            *(unsigned int*)&AsL[am * AS_STRIDE + pos] = pack2(l0, l1);
        }
        // ---------------- stage B into LDS in WMMA fragment order --------
        #pragma unroll
        for (int p = 0; p < 8; ++p) {
            const int kl = bk0 + 2 * p;
            const int pos = ((kl >> 3) & 1) * 16 + ((kl & 16) ? 8 : 0) + (kl & 7);
            __bf16 h0, l0, h1, l1;
            split_bf16(breg[2 * p], h0, l0);
            split_bf16(breg[2 * p + 1], h1, l1);
            *(unsigned int*)&BsH[bn * BS_STRIDE + pos] = pack2(h0, h1);
            *(unsigned int*)&BsL[bn * BS_STRIDE + pos] = pack2(l0, l1);
        }
        __syncthreads();

        // ---------------- WMMA compute: 2x2 subtiles, 3 terms each -------
        v16bf aH[2], aL[2], bH[2], bL[2];
        #pragma unroll
        for (int ms = 0; ms < 2; ++ms) {
            const int row = (wm * 32 + ms * 16 + l16) * AS_STRIDE + half * 16;
            aH[ms] = load_frag(&AsH[row]);
            aL[ms] = load_frag(&AsL[row]);
        }
        #pragma unroll
        for (int ns = 0; ns < 2; ++ns) {
            const int row = (wn * 32 + ns * 16 + l16) * BS_STRIDE + half * 16;
            bH[ns] = load_frag(&BsH[row]);
            bL[ns] = load_frag(&BsL[row]);
        }
        #pragma unroll
        for (int ms = 0; ms < 2; ++ms) {
            #pragma unroll
            for (int ns = 0; ns < 2; ++ns) {
                acc[ms][ns] = __builtin_amdgcn_wmma_f32_16x16x32_bf16(
                    false, aH[ms], false, bH[ns], (short)0, acc[ms][ns], false, false);
                acc[ms][ns] = __builtin_amdgcn_wmma_f32_16x16x32_bf16(
                    false, aH[ms], false, bL[ns], (short)0, acc[ms][ns], false, false);
                acc[ms][ns] = __builtin_amdgcn_wmma_f32_16x16x32_bf16(
                    false, aL[ms], false, bH[ns], (short)0, acc[ms][ns], false, false);
            }
        }
    }

    // ---------------- epilogue: bias + store (C layout: VGPR r -> M=r/r+8)
    #pragma unroll
    for (int ns = 0; ns < 2; ++ns) {
        const int ng = n0 + wn * 32 + ns * 16 + l16;
        const int b  = ng / HW;
        const int hw = ng - b * HW;
        float* op = out + (size_t)b * (COUT * HW) + hw;
        #pragma unroll
        for (int ms = 0; ms < 2; ++ms) {
            const int cobase = m0 + wm * 32 + ms * 16 + half * 8;
            #pragma unroll
            for (int r8 = 0; r8 < 8; ++r8) {
                const int co = cobase + r8;
                op[(size_t)co * HW] = acc[ms][ns][r8] + bias[co];
            }
        }
    }
}

// ---------------------------------------------------------------------------
// Sparsity: num_nonzero = sum_{ci,tap} Wcnt[ci,tap] * Xcnt[ci,tap]
// ---------------------------------------------------------------------------

__global__ void wcnt_kernel(const float* __restrict__ w, unsigned int* __restrict__ wcnt,
                            unsigned long long* __restrict__ acc) {
    const int idx = blockIdx.x * blockDim.x + threadIdx.x;   // 0..1151
    if (idx < CIN * 9) {
        const int ci = idx / 9;
        const int r  = idx - ci * 9;
        unsigned int c = 0;
        for (int co = 0; co < COUT; ++co)
            c += (w[co * KTOT + ci * 9 + r] != 0.0f) ? 1u : 0u;
        wcnt[idx] = c;
    }
    if (idx == 0) *acc = 0ull;
}

__global__ __launch_bounds__(256)
void nnz_kernel(const float* __restrict__ x, const unsigned int* __restrict__ wcnt,
                unsigned long long* __restrict__ acc) {
    __shared__ unsigned int red[8];
    unsigned int c = 0;
    const int idx = blockIdx.x * 256 + threadIdx.x;
    if (idx < 16 * CIN * HW) {
        const float v = x[idx];
        if (v != 0.0f) {
            const int plane = idx / HW;          // b*128 + ci
            const int ci = plane & (CIN - 1);
            const int hw = idx - plane * HW;
            const int h = hw / 56;
            const int w0 = hw - h * 56;
            const unsigned int* wc = wcnt + ci * 9;
            #pragma unroll
            for (int kh = 0; kh < 3; ++kh) {
                const int oh = h - kh + 1;
                if ((unsigned)oh < 56u) {
                    #pragma unroll
                    for (int kw = 0; kw < 3; ++kw) {
                        const int ow = w0 - kw + 1;
                        if ((unsigned)ow < 56u) c += wc[kh * 3 + kw];
                    }
                }
            }
        }
    }
    #pragma unroll
    for (int off = 16; off > 0; off >>= 1) c += __shfl_down(c, off, 32);
    if ((threadIdx.x & 31) == 0) red[threadIdx.x >> 5] = c;
    __syncthreads();
    if (threadIdx.x == 0) {
        unsigned int s = 0;
        #pragma unroll
        for (int i = 0; i < 8; ++i) s += red[i];
        atomicAdd(acc, (unsigned long long)s);
    }
}

__global__ void fin_kernel(const unsigned long long* __restrict__ acc,
                           float* __restrict__ out_sp) {
    const double total = 14797504512.0;  // 16*56*56*256*128*9
    const double nnz = (double)(*acc);
    *out_sp = (float)((total - nnz) / total);
}

// ---------------------------------------------------------------------------

extern "C" void kernel_launch(void* const* d_in, const int* in_sizes, int n_in,
                              void* d_out, int out_size, void* d_ws, size_t ws_size,
                              hipStream_t stream) {
    const float* x    = (const float*)d_in[0];
    const float* w    = (const float*)d_in[1];
    const float* bias = (const float*)d_in[2];
    float* out = (float*)d_out;

    unsigned int* wcnt = (unsigned int*)d_ws;
    unsigned long long* acc =
        (unsigned long long*)((char*)d_ws + (size_t)CIN * 9 * sizeof(unsigned int));

    // Main conv: grid = N tiles (50176/128=392) x M tiles (256/64=4)
    dim3 grid(392, 4);
    conv_wmma_kernel<<<grid, 256, 0, stream>>>(x, w, bias, out);

    // Sparsity pipeline (ordered on the same stream)
    wcnt_kernel<<<5, 256, 0, stream>>>(w, wcnt, acc);
    nnz_kernel<<<(16 * CIN * HW) / 256, 256, 0, stream>>>(x, wcnt, acc);
    fin_kernel<<<1, 1, 0, stream>>>(acc, out + (size_t)16 * COUT * HW);
}
